// SparseLinear_80547816669709
// MI455X (gfx1250) — matrix operand order repaired and checked by reading, not compiled, not run on hardware
//
#include <hip/hip_runtime.h>
#include <stdint.h>

// Problem constants (match reference)
#define BATCH   16384
#define IN_F    4096
#define OUT_F   4096
#define STRIPS  256          // OUT_F / 16
#define CAP     4096         // per-strip nnz capacity (mean 3277, +14 sigma)
#define CHUNKS_PER_STRIP 128 // CAP / 32
#define TOTAL_CHUNKS (STRIPS * CHUNKS_PER_STRIP)
#define META_BYTES 1152      // per-chunk: 1024B A-image + 128B col-offsets

typedef __attribute__((ext_vector_type(16))) __bf16          v16bf;
typedef __attribute__((ext_vector_type(8)))  float           v8f;
typedef __attribute__((ext_vector_type(8)))  unsigned int    v8u;
typedef __attribute__((ext_vector_type(4)))  unsigned int    u32x4;
typedef __attribute__((ext_vector_type(8)))  int             i32x8;
typedef __attribute__((ext_vector_type(4)))  int             i32x4;

__device__ __forceinline__ unsigned short f32_to_bf16_rne(float f) {
    unsigned int u = __builtin_bit_cast(unsigned int, f);
    u += 0x7FFFu + ((u >> 16) & 1u);
    return (unsigned short)(u >> 16);
}

// ---------------------------------------------------------------------------
// Phase 0: zero the strip counters + strip data (padding entries must be 0)
// ---------------------------------------------------------------------------
__global__ void zero_kernel(uint4* p, long n16) {
    long i = (long)blockIdx.x * blockDim.x + threadIdx.x;
    if (i < n16) p[i] = make_uint4(0u, 0u, 0u, 0u);
}

// ---------------------------------------------------------------------------
// Phase 1: scatter COO nnz into 16-row strips. Entry = val_bf16<<16|lrow<<12|col
// ---------------------------------------------------------------------------
__global__ void scatter_kernel(const int* __restrict__ idx,
                               const float* __restrict__ val, int nnz,
                               unsigned int* __restrict__ cnt,
                               unsigned int* __restrict__ sdata) {
    int i = blockIdx.x * blockDim.x + threadIdx.x;
    if (i >= nnz) return;
    unsigned int r = (unsigned int)idx[2 * i];
    unsigned int c = (unsigned int)idx[2 * i + 1];
    unsigned int s = r >> 4, lr = r & 15u;
    unsigned int pos = atomicAdd(&cnt[s], 1u);
    if (pos < CAP) {
        unsigned int v = (unsigned int)f32_to_bf16_rne(val[i]);
        sdata[(size_t)s * CAP + pos] = (v << 16) | (lr << 12) | (c & 0xFFFu);
    }
}

// ---------------------------------------------------------------------------
// Phase 2: transpose x [BATCH][IN] f32 -> xT [IN][BATCH] bf16 (LDS tiled).
// bf16 xT = 128 MB -> resident in the 192 MB L2 for the gather phase.
// ---------------------------------------------------------------------------
__global__ void transpose_bf16_kernel(const float* __restrict__ x,
                                      unsigned short* __restrict__ xT) {
    __shared__ float tile[32][33];
    int c0 = blockIdx.x * 32, b0 = blockIdx.y * 32;
    int tx = threadIdx.x, ty = threadIdx.y;
#pragma unroll
    for (int i = 0; i < 4; i++)
        tile[ty + 8 * i][tx] = x[(size_t)(b0 + ty + 8 * i) * IN_F + c0 + tx];
    __syncthreads();
#pragma unroll
    for (int i = 0; i < 4; i++)
        xT[(size_t)(c0 + ty + 8 * i) * BATCH + b0 + tx] =
            f32_to_bf16_rne(tile[tx][ty + 8 * i]);
}

// ---------------------------------------------------------------------------
// Phase 3: per 32-nnz chunk, pre-swizzle the 16x32 bf16 A-operand into its
// exact per-lane VGPR image (ISA 7.12.2: lane m holds row m; half h of the
// 16-half vector maps to K = h + 8*(h>>3) + 8*(lane>=16)). A-image (1024B) and
// the pre-shifted column byte-offsets (c << 15, 128B) are packed contiguously
// per chunk so one TDM descriptor can stage both.
// ---------------------------------------------------------------------------
__global__ void pack_kernel(const unsigned int* __restrict__ cnt,
                            const unsigned int* __restrict__ sdata,
                            unsigned int* __restrict__ meta) {
    const int lane = threadIdx.x & 31;
    unsigned int g = blockIdx.x * 8u + (unsigned int)(threadIdx.x >> 5);
    unsigned int s = g >> 7;            // /CHUNKS_PER_STRIP
    unsigned int ck = g & 127u;
    unsigned int n = cnt[s]; if (n > CAP) n = CAP;
    if (ck * 32u >= n) return;          // wave-uniform
    unsigned int* mb = meta + (size_t)g * (META_BYTES / 4);
    unsigned int e = sdata[(size_t)g * 32 + lane];
    mb[256 + lane] = (e & 0xFFFu) << 15;          // byte offset into xT
    unsigned int m   = lane & 15u;
    unsigned int hi8 = (lane >> 4) << 3;          // 0 or 8
    unsigned int dw[8];
#pragma unroll
    for (int j = 0; j < 8; j++) {
        unsigned int a01[2];
#pragma unroll
        for (int q = 0; q < 2; q++) {
            int h = 2 * j + q;
            int K = h + 8 * (h >> 3) + (int)hi8;  // A-operand K map
            unsigned int eh = (unsigned int)__shfl((int)e, K);
            a01[q] = (((eh >> 12) & 15u) == m) ? (eh >> 16) : 0u;
        }
        dw[j] = a01[0] | (a01[1] << 16);
    }
    uint4* o = (uint4*)(mb + (size_t)lane * 8);
    o[0] = make_uint4(dw[0], dw[1], dw[2], dw[3]);
    o[1] = make_uint4(dw[4], dw[5], dw[6], dw[7]);
}

// ---------------------------------------------------------------------------
// TDM: stage one chunk's 1152B meta block (global -> LDS) with a single
// TENSOR_LOAD_TO_LDS. D# group0: count=1 | lds_addr | 57-bit global_addr |
// type=2. Group1: data_size=1B, tensor_dim0=tile_dim0=1152, tile_dim1=1,
// tensor_dim0_stride=1152. Groups 2/3 zero (<=2D tile). Tracked by TENSORcnt.
// ---------------------------------------------------------------------------
#if __has_builtin(__builtin_amdgcn_tensor_load_to_lds)
#define HAVE_TDM 1
__device__ __forceinline__ void tdm_stage_meta(unsigned long long ga,
                                               unsigned int ldsaddr) {
    u32x4 g0;
    g0[0] = 1u;                                          // count=1, user D#
    g0[1] = ldsaddr;                                     // lds_addr (bytes)
    g0[2] = (unsigned int)(ga & 0xFFFFFFFFull);          // global_addr[31:0]
    g0[3] = (unsigned int)((ga >> 32) & 0x1FFFFFFull)    // global_addr[56:32]
            | (2u << 30);                                // type=2 ("image")
    i32x8 g1;
    g1[0] = 0;                                           // wg_mask=0, data_size=1B
    g1[1] = (int)((unsigned)META_BYTES << 16);           // tensor_dim0[15:0]
    g1[2] = (int)(1u << 16);                             // tensor_dim1[15:0]=1
    g1[3] = (int)((unsigned)META_BYTES << 16);           // tile_dim0=1152
    g1[4] = 1;                                           // tile_dim1=1
    g1[5] = META_BYTES;                                  // tensor_dim0_stride
    g1[6] = 0;
    g1[7] = 0;
    i32x4 z4 = {0, 0, 0, 0};
#if defined(__clang_major__) && (__clang_major__ >= 23)
    i32x8 z8 = {0, 0, 0, 0, 0, 0, 0, 0};
    __builtin_amdgcn_tensor_load_to_lds(g0, g1, z4, z4, z8, 0);
#else
    __builtin_amdgcn_tensor_load_to_lds(g0, g1, z4, z4, 0);
#endif
}
#else
#define HAVE_TDM 0
#endif

__device__ __forceinline__ void stage_meta(unsigned long long mBase,
                                           unsigned int gq, unsigned int dst,
                                           int lane) {
#if HAVE_TDM
    (void)lane;
    tdm_stage_meta(mBase + (unsigned long long)gq * META_BYTES, dst);
#else
    unsigned int l16 = (unsigned int)lane * 16u;
    unsigned int go  = gq * (unsigned int)META_BYTES;
    asm volatile("global_load_async_to_lds_b128 %0, %1, %2"
                 :: "v"(dst + l16), "v"(go + l16), "s"(mBase) : "memory");
    asm volatile("global_load_async_to_lds_b128 %0, %1, %2"
                 :: "v"(dst + 512u + l16), "v"(go + 512u + l16), "s"(mBase) : "memory");
    asm volatile("global_load_async_to_lds_b32 %0, %1, %2"
                 :: "v"(dst + 1024u + (unsigned)lane * 4u),
                    "v"(go + 1024u + (unsigned)lane * 4u), "s"(mBase) : "memory");
#endif
}

__device__ __forceinline__ void wait_stage() {
#if HAVE_TDM
    __builtin_amdgcn_s_wait_tensorcnt(0);
#else
    asm volatile("s_wait_asynccnt 0" ::: "memory");
#endif
}

// ---------------------------------------------------------------------------
// Phase 4: WMMA SpMM. Block = 8 waves; wave = 16 features x 128 batch (8 tiles).
// Batch col N of tile t -> global batch = base + 8*N + t, so one b128 gather
// feeds all eight B tiles at one K slot; the half-transpose is one v_perm_b32
// per B-operand dword. Per-chunk meta is staged once per WG into LDS by the
// TDM (TENSORcnt), double-buffered across chunks. __launch_bounds__(256,1)
// relaxes the VGPR target so each B tile keeps its own register range
// (removes the WMMA WAR-hazard v_nop padding seen at default occupancy).
// ---------------------------------------------------------------------------
__global__ void __launch_bounds__(256, 1)
spmm_wmma_kernel(const unsigned short* __restrict__ xT,
                 const unsigned int* __restrict__ cnt,
                 const unsigned int* __restrict__ meta,
                 const float* __restrict__ bias,
                 float* __restrict__ out) {
    // 2 x (1024B A-image + 128B col-offsets + 384B pad)
    __shared__ __align__(32) unsigned int chbuf[2][384];

    const int lane  = threadIdx.x & 31;
    const int w     = threadIdx.x >> 5;
    const int s     = blockIdx.y;
    const int bbase = blockIdx.x * 1024 + w * 128;
    const int laneN = lane & 15;
    const int hiSel = lane & 16;       // B-operand K-half select
    const int half  = lane >> 4;       // D: this lane's VGPRs hold M = r + 8*half

    unsigned int n = cnt[s]; if (n > CAP) n = CAP;
    const int nch = (int)((n + 31u) >> 5);

    v8f z = {0.f, 0.f, 0.f, 0.f, 0.f, 0.f, 0.f, 0.f};
    v8f acc[8];
#pragma unroll
    for (int t = 0; t < 8; t++) acc[t] = z;

    const char* xTb = (const char*)xT;
    const unsigned int laneOff = (unsigned int)((bbase + 8 * laneN) * 2);
    const unsigned long long mBase = (unsigned long long)(const void*)meta;
    const unsigned int g0   = (unsigned int)s * 128u;
    const unsigned int lds0 = (unsigned int)(uintptr_t)&chbuf[0][0];
    const unsigned int lds1 = (unsigned int)(uintptr_t)&chbuf[1][0];

    // Prologue: wave 0 launches the stage of chunk 0.
    if (w == 0 && nch > 0) stage_meta(mBase, g0, lds0, lane);

    for (int ch = 0; ch < nch; ch++) {
        const int p = ch & 1;
        // Chunk ch's staged copy (issued last iteration) must have landed.
        if (w == 0) wait_stage();
        __syncthreads();
        // Overlap: stage chunk ch+1 into the other buffer while computing ch.
        if (w == 0 && ch + 1 < nch)
            stage_meta(mBase, g0 + (unsigned int)(ch + 1), p ? lds0 : lds1, lane);

        v8u araw = *(const v8u*)&chbuf[p][lane * 8];   // 2x ds_load_b128
        v16bf A = __builtin_bit_cast(v16bf, araw);

        v8u bt[8];
#pragma unroll
        for (int j = 0; j < 8; j++) {
            // Two adjacent K-slot byte-offsets for this lane half: one ds_load_b64
            uint2 cp = *(const uint2*)&chbuf[p][256 + (hiSel >> 1) + 2 * j + 8 * (j >> 2)];
            uint4 ldA = *(const uint4*)(xTb + (cp.x + laneOff));  // 8 batch cols @ K=2j
            uint4 ldB = *(const uint4*)(xTb + (cp.y + laneOff));  // 8 batch cols @ K=2j+1
            unsigned int la[4] = {ldA.x, ldA.y, ldA.z, ldA.w};
            unsigned int lb[4] = {ldB.x, ldB.y, ldB.z, ldB.w};
#pragma unroll
            for (int q = 0; q < 4; q++) {
                // dword q holds batch 2q (lo half) and 2q+1 (hi half)
                bt[2 * q][j]     = __builtin_amdgcn_perm(lb[q], la[q], 0x05040100u);
                bt[2 * q + 1][j] = __builtin_amdgcn_perm(lb[q], la[q], 0x07060302u);
            }
        }
#pragma unroll
        for (int t = 0; t < 8; t++)
            acc[t] = __builtin_amdgcn_wmma_f32_16x16x32_bf16(false, A, false,
                       __builtin_bit_cast(v16bf, bt[t]), (short)0, acc[t], false, false);
    }

    // D layout: lane holds 8 consecutive features M = 8*half..8*half+7 for one
    // batch row -> directly contiguous in out[batch][feature]. Add bias, store.
    const float* bp = bias + s * 16 + 8 * half;
    float4 bv0 = *(const float4*)(bp);
    float4 bv1 = *(const float4*)(bp + 4);
#pragma unroll
    for (int t = 0; t < 8; t++) {
        float* op = out + (size_t)(bbase + 8 * laneN + t) * OUT_F + s * 16 + 8 * half;
        float4 o0 = make_float4(acc[t][0] + bv0.x, acc[t][1] + bv0.y,
                                acc[t][2] + bv0.z, acc[t][3] + bv0.w);
        float4 o1 = make_float4(acc[t][4] + bv1.x, acc[t][5] + bv1.y,
                                acc[t][6] + bv1.z, acc[t][7] + bv1.w);
        *(float4*)(op)     = o0;
        *(float4*)(op + 4) = o1;
    }
}

// ---------------------------------------------------------------------------
// Workspace layout (bytes):
//   [0,     4096)       strip counters (256 u32, padded)
//   [4096,  +4M)        strip-sorted packed COO
//   [..,    +36M)       per-chunk meta blocks (32768 * 1152B: A-image + cols)
//   [..,    +128M)      xT bf16 [IN][BATCH]
// Total ~176 MB.
// ---------------------------------------------------------------------------
extern "C" void kernel_launch(void* const* d_in, const int* in_sizes, int n_in,
                              void* d_out, int out_size, void* d_ws, size_t ws_size,
                              hipStream_t stream) {
    const float* x    = (const float*)d_in[0];
    const float* wval = (const float*)d_in[1];
    const int*   widx = (const int*)d_in[2];
    const float* bias = (const float*)d_in[3];
    const int nnz = in_sizes[1];

    char* ws = (char*)d_ws;
    unsigned int*   cnt   = (unsigned int*)(ws);
    unsigned int*   sdata = (unsigned int*)(ws + 4096);
    unsigned int*   meta  = (unsigned int*)(ws + 4096 + 4194304ull);
    unsigned short* xT    = (unsigned short*)(ws + 4096 + 4194304ull
                                  + (unsigned long long)TOTAL_CHUNKS * META_BYTES);

    const long zero16 = (4096 + 4194304l) / 16;
    zero_kernel<<<(unsigned)((zero16 + 255) / 256), 256, 0, stream>>>((uint4*)ws, zero16);

    scatter_kernel<<<(nnz + 255) / 256, 256, 0, stream>>>(widx, wval, nnz, cnt, sdata);

    transpose_bf16_kernel<<<dim3(IN_F / 32, BATCH / 32), dim3(32, 8), 0, stream>>>(x, xT);

    pack_kernel<<<TOTAL_CHUNKS / 8, 256, 0, stream>>>(cnt, sdata, meta);

    spmm_wmma_kernel<<<dim3(BATCH / 1024, STRIPS), 256, 0, stream>>>(
        xT, cnt, meta, bias, (float*)d_out);
}
